// FCOSRPN_42288247996676
// MI455X (gfx1250) — matrix operand order repaired and checked by reading, not compiled, not run on hardware
//
#include <hip/hip_runtime.h>

// ---------------------------------------------------------------------------
// FCOS head for MI455X (gfx1250): bf16 WMMA implicit-GEMM 3x3 convs
// (V_WMMA_F32_16X16X32_BF16), LDS-staged activations, packed [tap][co][ci]
// bf16 weights, software-pipelined B-fragment loads, + GroupNorm / decode /
// NMS side kernels.
// ---------------------------------------------------------------------------

typedef __attribute__((ext_vector_type(16))) __bf16 v16bf;
typedef __attribute__((ext_vector_type(8)))  float  v8f;
typedef __attribute__((ext_vector_type(4)))  int    i32x4;

union Frag16 { v16bf bf; i32x4 q[2]; };

#define CI_CH 256

__device__ __forceinline__ unsigned short f2bf(float f) {
  unsigned u = __float_as_uint(f);
  u += 0x7FFFu + ((u >> 16) & 1u);          // round-to-nearest-even
  return (unsigned short)(u >> 16);
}
__device__ __forceinline__ float sigmoidf_(float x) { return 1.f / (1.f + __expf(-x)); }

// ---------------------------------------------------------------------------
// Implicit-GEMM 3x3 conv, pad=1, stride=1.  N=2, CI=256, COpad = NPASS*64.
// Input  : f32 NCHW.     Weights: bf16 packed [9][COpad][256] (tap, co, ci).
// Output : f32 [N][COpad][H][W], only co < CO_store stored (bias added).
// Block = 256 threads (8 wave32), tile = 2 rows x 64 cols of output pixels.
//
// Pipelining: all 4 A fragments of a tap are preloaded from LDS; B fragments
// are double-buffered in groups of 4 (next co-pass loads issued before the
// current pass's WMMAs) so the matrix pipe is not stalled on s_wait_loadcnt 0.
// ---------------------------------------------------------------------------
template <int NPASS>
__global__ __launch_bounds__(256) void conv3x3_wmma_kernel(
    const float* __restrict__ in, const unsigned short* __restrict__ wp,
    const float* __restrict__ bias, float* __restrict__ out,
    int H, int W, int CO_store) {
  constexpr int COpad = NPASS * 64;
  // slab[yy 0..3][xx 0..65][ci 0..127 (+8 pad)]  bf16  -> 71808 bytes of LDS
  __shared__ unsigned short slab[4 * 66 * 136];

  const int tid = threadIdx.x;
  const int lane = tid & 31;
  const int wave = tid >> 5;
  const int khalf = lane >> 4;
  const int lrow = lane & 15;
  const int row_off = wave >> 2;        // 0..1
  const int x_off = (wave & 3) << 4;    // 0,16,32,48

  const int gxt = (W + 63) >> 6;
  const int gyt = (H + 1) >> 1;
  int b = blockIdx.x;
  const int xt = b % gxt; b /= gxt;
  const int yt = b % gyt;
  const int n = b / gyt;
  const int x0 = xt << 6, y0 = yt << 1;

  __builtin_prefetch(wp + (size_t)tid * 128, 0, 0);  // global_prefetch_b8

  v8f acc[NPASS][4];
#pragma unroll
  for (int p = 0; p < NPASS; ++p)
#pragma unroll
    for (int j = 0; j < 4; ++j) acc[p][j] = (v8f)0.0f;

  const size_t hw = (size_t)H * W;

  for (int hf = 0; hf < 2; ++hf) {      // two 128-channel halves of CI
    __syncthreads();
    // cooperative fill: global f32 -> bf16 LDS slab, zero-pad at borders
    for (int i = tid; i < 128 * 4 * 66; i += 256) {
      int xx = i % 66;
      int t2 = i / 66;
      int yy = t2 & 3;
      int ci = t2 >> 2;
      int gy = y0 - 1 + yy, gx = x0 - 1 + xx;
      float v = 0.f;
      if ((unsigned)gy < (unsigned)H && (unsigned)gx < (unsigned)W)
        v = in[(size_t)(n * CI_CH + hf * 128 + ci) * hw + (size_t)gy * W + gx];
      slab[(yy * 66 + xx) * 136 + ci] = f2bf(v);
    }
    __syncthreads();

    for (int t = 0; t < 9; ++t) {       // 3x3 taps
      const int dy = t / 3, dx = t - dy * 3;
      const int yy = row_off + dy;
      const unsigned short* wt = wp + (size_t)t * COpad * CI_CH + hf * 128;

      // preload all 4 A fragments (16x32 bf16 each): 8 ds_load_b128
      Frag16 afr[4];
      const unsigned short* abase =
          &slab[(yy * 66 + (x_off + lrow + dx)) * 136 + khalf * 8];
#pragma unroll
      for (int cib = 0; cib < 4; ++cib) {
        afr[cib].q[0] = *(const i32x4*)(abase + cib * 32);
        afr[cib].q[1] = *(const i32x4*)(abase + cib * 32 + 16);
      }

#pragma unroll
      for (int cib = 0; cib < 4; ++cib) {  // 32-ci K blocks within the half
        const unsigned short* wb = wt + cib * 32 + khalf * 8;

        // B fragments, double-buffered in groups of 4 (one co-pass)
        Frag16 bq[2][4];
#pragma unroll
        for (int j = 0; j < 4; ++j) {
          const unsigned short* bp = wb + (size_t)(j * 16 + lrow) * CI_CH;
          bq[0][j].q[0] = *(const i32x4*)bp;
          bq[0][j].q[1] = *(const i32x4*)(bp + 16);
        }
#pragma unroll
        for (int p = 0; p < NPASS; ++p) {
          if (p + 1 < NPASS) {           // issue next group's loads first
#pragma unroll
            for (int j = 0; j < 4; ++j) {
              const unsigned short* bp =
                  wb + (size_t)((p + 1) * 64 + j * 16 + lrow) * CI_CH;
              bq[(p + 1) & 1][j].q[0] = *(const i32x4*)bp;
              bq[(p + 1) & 1][j].q[1] = *(const i32x4*)(bp + 16);
            }
          }
#pragma unroll
          for (int j = 0; j < 4; ++j)
            acc[p][j] = __builtin_amdgcn_wmma_f32_16x16x32_bf16(
                false, afr[cib].bf, false, bq[p & 1][j].bf, (short)0,
                acc[p][j], false, false);
        }
      }
    }
  }

  // store: C-matrix VGPR r -> M = r + 8*khalf (pixel), N = lrow (co)
  const int y = y0 + row_off;
  if (y < H) {
#pragma unroll
    for (int p = 0; p < NPASS; ++p)
#pragma unroll
      for (int j = 0; j < 4; ++j) {
        const int co = p * 64 + j * 16 + lrow;
        if (co < CO_store) {
          const float bz = bias[co];
          float* op = out + ((size_t)n * COpad + co) * hw + (size_t)y * W;
#pragma unroll
          for (int r = 0; r < 8; ++r) {
            int x = x0 + x_off + r + khalf * 8;
            if (x < W) op[x] = acc[p][j][r] + bz;
          }
        }
      }
  }
}

// ---------------------------------------------------------------------------
// Weight packing: f32 OIHW -> bf16 [tap][co][ci]
// ---------------------------------------------------------------------------
__global__ void pack_tower_w(const float* __restrict__ w, unsigned short* __restrict__ wp) {
  // w: [4][256][256][3][3] -> wp: [4][9][256][256]
  const int total = 4 * 9 * 256 * 256;
  for (int i = blockIdx.x * 256 + threadIdx.x; i < total; i += gridDim.x * 256) {
    int ci = i & 255;
    int r = i >> 8;
    int co = r & 255; r >>= 8;
    int t = r % 9;
    int l = r / 9;
    int ty = t / 3, tx = t - ty * 3;
    float v = w[((((size_t)l * 256 + co) * 256 + ci) * 3 + ty) * 3 + tx];
    wp[i] = f2bf(v);
  }
}

__global__ void pack_headA_w(const float* __restrict__ score_w, const float* __restrict__ ctr_w,
                             unsigned short* __restrict__ wp) {
  // -> [9][128][256]: co<80 score, co==80 ctr, rest 0
  const int total = 9 * 128 * 256;
  for (int i = blockIdx.x * 256 + threadIdx.x; i < total; i += gridDim.x * 256) {
    int ci = i & 255;
    int r = i >> 8;
    int co = r & 127;
    int t = r >> 7;
    int ty = t / 3, tx = t - ty * 3;
    float v = 0.f;
    if (co < 80)       v = score_w[(((size_t)co * 256 + ci) * 3 + ty) * 3 + tx];
    else if (co == 80) v = ctr_w[(((size_t)ci) * 3 + ty) * 3 + tx];
    wp[i] = f2bf(v);
  }
}

__global__ void pack_headB_w(const float* __restrict__ bbox_w, unsigned short* __restrict__ wp) {
  // -> [9][64][256]: co<4 bbox, rest 0
  const int total = 9 * 64 * 256;
  for (int i = blockIdx.x * 256 + threadIdx.x; i < total; i += gridDim.x * 256) {
    int ci = i & 255;
    int r = i >> 8;
    int co = r & 63;
    int t = r >> 6;
    int ty = t / 3, tx = t - ty * 3;
    float v = (co < 4) ? bbox_w[(((size_t)co * 256 + ci) * 3 + ty) * 3 + tx] : 0.f;
    wp[i] = f2bf(v);
  }
}

// ---------------------------------------------------------------------------
// Per-call init: zero candidates/counters/output, build packed head biases
// ---------------------------------------------------------------------------
__global__ void init_kernel(float* cand, int* cnt, float* out, float* biasA, float* biasB,
                            const float* score_b, const float* ctr_b, const float* bbox_b) {
  int i = blockIdx.x * 256 + threadIdx.x;
  if (i < 2 * 5000 * 6) cand[i] = 0.f;
  if (i < 1200) out[i] = 0.f;
  if (i < 10) cnt[i] = 0;
  if (i < 128) biasA[i] = (i < 80) ? score_b[i] : (i == 80 ? ctr_b[0] : 0.f);
  if (i < 64) biasB[i] = (i < 4) ? bbox_b[i] : 0.f;
}

// ---------------------------------------------------------------------------
// GroupNorm: stats then apply+ReLU. Group = 8 contiguous channels.
// ---------------------------------------------------------------------------
__global__ void gn_stats_kernel(const float* __restrict__ x, float* __restrict__ stats,
                                int H, int W) {
  const int blk = blockIdx.x;               // n*32 + g
  const size_t cnt = (size_t)8 * H * W;     // contiguous region
  const float* p = x + (size_t)blk * cnt;
  float s1 = 0.f, s2 = 0.f;
  for (size_t i = threadIdx.x; i < cnt; i += 256) {
    float v = p[i];
    s1 += v;
    s2 += v * v;
  }
  __shared__ float a1[256], a2[256];
  a1[threadIdx.x] = s1;
  a2[threadIdx.x] = s2;
  __syncthreads();
  for (int s = 128; s > 0; s >>= 1) {
    if ((int)threadIdx.x < s) {
      a1[threadIdx.x] += a1[threadIdx.x + s];
      a2[threadIdx.x] += a2[threadIdx.x + s];
    }
    __syncthreads();
  }
  if (threadIdx.x == 0) {
    float mean = a1[0] / (float)cnt;
    float var = a2[0] / (float)cnt - mean * mean;
    stats[blk * 2] = mean;
    stats[blk * 2 + 1] = rsqrtf(var + 1e-5f);
  }
}

__global__ void gn_apply_relu_kernel(const float* __restrict__ x, float* __restrict__ y,
                                     const float* __restrict__ stats,
                                     const float* __restrict__ s, const float* __restrict__ b,
                                     int H, int W) {
  const size_t hw = (size_t)H * W;
  const size_t total = (size_t)2 * 256 * hw;
  for (size_t i = blockIdx.x * 256ull + threadIdx.x; i < total; i += (size_t)gridDim.x * 256) {
    int c = (int)((i / hw) & 255);
    int n = (int)(i / (hw * 256));
    int g = c >> 3;
    float mean = stats[(n * 32 + g) * 2];
    float rstd = stats[(n * 32 + g) * 2 + 1];
    float v = (x[i] - mean) * rstd * s[c] + b[c];
    y[i] = fmaxf(v, 0.f);
  }
}

// ---------------------------------------------------------------------------
// Decode one level: scores = sqrt(sig(logit)*sig(ctr)); boxes from regressed
// distances; append above-threshold candidates (capped 1000/level ~ top-k).
// ---------------------------------------------------------------------------
__global__ void decode_kernel(const float* __restrict__ ha, const float* __restrict__ hb,
                              const float* __restrict__ scales, int level, int H, int W,
                              float stride, float* __restrict__ cand, int* __restrict__ cnt) {
  const int hw = H * W;
  const int total = 2 * hw;
  int i = blockIdx.x * 256 + threadIdx.x;
  if (i >= total) return;
  int n = i / hw, p = i - n * hw;
  int y = p / W, x = p - y * W;
  size_t baseA = (size_t)n * 128 * hw + p;
  size_t baseB = (size_t)n * 64 * hw + p;
  float sctr = sigmoidf_(ha[baseA + (size_t)80 * hw]);
  float sc5 = scales[level];
  float r0 = fmaxf(sc5 * hb[baseB], 0.f);
  float r1 = fmaxf(sc5 * hb[baseB + (size_t)hw], 0.f);
  float r2 = fmaxf(sc5 * hb[baseB + (size_t)2 * hw], 0.f);
  float r3 = fmaxf(sc5 * hb[baseB + (size_t)3 * hw], 0.f);
  float px = (x + 0.5f) * stride, py = (y + 0.5f) * stride;
  float x1 = px - r0 * stride, y1 = py - r1 * stride;
  float x2 = px + r2 * stride, y2 = py + r3 * stride;
  for (int k = 0; k < 80; ++k) {
    float sc = sqrtf(sigmoidf_(ha[baseA + (size_t)k * hw]) * sctr);
    if (sc > 0.05f) {
      int idx = atomicAdd(&cnt[n * 5 + level], 1);
      if (idx < 1000) {
        float* c6 = cand + ((size_t)(n * 5 + level) * 1000 + idx) * 6;
        c6[0] = x1; c6[1] = y1; c6[2] = x2; c6[3] = y2;
        c6[4] = sc; c6[5] = (float)k;
      }
    }
  }
}

// ---------------------------------------------------------------------------
// Class-aware NMS, 100 rounds, one block per image over 5000 candidates.
// ---------------------------------------------------------------------------
__global__ void nms_kernel(const float* __restrict__ cand, float* __restrict__ out) {
  const int n = blockIdx.x;
  const int tid = threadIdx.x;
  const int M = 5000;
  __shared__ float sc[5000];
  __shared__ float rv[256];
  __shared__ int ri[256];
  __shared__ float chosen[6];
  __shared__ int sel;
  const float* cn = cand + (size_t)n * M * 6;
  for (int i = tid; i < M; i += 256) sc[i] = cn[i * 6 + 4];
  __syncthreads();
  for (int it = 0; it < 100; ++it) {
    float bv = -1.f;
    int bi = 0;
    for (int i = tid; i < M; i += 256) {
      float v = sc[i];
      if (v > bv) { bv = v; bi = i; }
    }
    rv[tid] = bv;
    ri[tid] = bi;
    __syncthreads();
    for (int s = 128; s > 0; s >>= 1) {
      if (tid < s && rv[tid + s] > rv[tid]) { rv[tid] = rv[tid + s]; ri[tid] = ri[tid + s]; }
      __syncthreads();
    }
    if (tid == 0) {
      if (rv[0] > 0.f) {
        int i = ri[0];
        sel = i;
        for (int c = 0; c < 6; ++c) chosen[c] = cn[i * 6 + c];
        float* o = out + ((size_t)n * 100 + it) * 6;
        o[0] = chosen[0]; o[1] = chosen[1]; o[2] = chosen[2]; o[3] = chosen[3];
        o[4] = rv[0];     o[5] = chosen[5];
      } else {
        sel = -1;
      }
    }
    __syncthreads();
    if (sel >= 0) {
      float off = chosen[5] * 1e4f;
      float cx1 = chosen[0] + off, cy1 = chosen[1] + off;
      float cx2 = chosen[2] + off, cy2 = chosen[3] + off;
      float ca = fmaxf(cx2 - cx1, 0.f) * fmaxf(cy2 - cy1, 0.f);
      for (int i = tid; i < M; i += 256) {
        if (sc[i] <= 0.f) continue;
        float o2 = cn[i * 6 + 5] * 1e4f;
        float x1 = cn[i * 6 + 0] + o2, y1 = cn[i * 6 + 1] + o2;
        float x2 = cn[i * 6 + 2] + o2, y2 = cn[i * 6 + 3] + o2;
        float iw = fmaxf(fminf(cx2, x2) - fmaxf(cx1, x1), 0.f);
        float ih = fmaxf(fminf(cy2, y2) - fmaxf(cy1, y1), 0.f);
        float inter = iw * ih;
        float a2 = fmaxf(x2 - x1, 0.f) * fmaxf(y2 - y1, 0.f);
        float iou = inter / fmaxf(ca + a2 - inter, 1e-6f);
        if (iou > 0.6f || i == sel) sc[i] = 0.f;
      }
    }
    __syncthreads();
  }
}

// ---------------------------------------------------------------------------
// Host orchestration
// ---------------------------------------------------------------------------
extern "C" void kernel_launch(void* const* d_in, const int* in_sizes, int n_in,
                              void* d_out, int out_size, void* d_ws, size_t ws_size,
                              hipStream_t stream) {
  (void)in_sizes; (void)n_in; (void)out_size; (void)ws_size;

  static const int Hs[5] = {100, 50, 25, 13, 7};
  static const int Ws[5] = {152, 76, 38, 19, 10};
  static const float Sv[5] = {8.f, 16.f, 32.f, 64.f, 128.f};

  const float* feats[5] = {(const float*)d_in[0], (const float*)d_in[1], (const float*)d_in[2],
                           (const float*)d_in[3], (const float*)d_in[4]};
  const float* cls_w = (const float*)d_in[5];
  const float* cls_b = (const float*)d_in[6];
  const float* cls_gn_s = (const float*)d_in[7];
  const float* cls_gn_b = (const float*)d_in[8];
  const float* box_w = (const float*)d_in[9];
  const float* box_b = (const float*)d_in[10];
  const float* box_gn_s = (const float*)d_in[11];
  const float* box_gn_b = (const float*)d_in[12];
  const float* score_w = (const float*)d_in[13];
  const float* score_b = (const float*)d_in[14];
  const float* bbox_w = (const float*)d_in[15];
  const float* bbox_b = (const float*)d_in[16];
  const float* ctr_w = (const float*)d_in[17];
  const float* ctr_b = (const float*)d_in[18];
  const float* scales = (const float*)d_in[19];
  float* out = (float*)d_out;

  // workspace carve-up (sized for level 0)
  char* base = (char*)d_ws;
  size_t off = 0;
  auto alloc = [&](size_t bytes) -> void* {
    void* p = base + off;
    off = (off + bytes + 255) & ~(size_t)255;
    return p;
  };
  const size_t hw0 = (size_t)100 * 152;
  float* P0 = (float*)alloc(2 * 256 * hw0 * 4);
  float* P1 = (float*)alloc(2 * 256 * hw0 * 4);
  float* HA = (float*)alloc(2 * 128 * hw0 * 4);
  float* HB = (float*)alloc(2 * 64 * hw0 * 4);
  unsigned short* WpC = (unsigned short*)alloc((size_t)4 * 9 * 256 * 256 * 2);
  unsigned short* WpB = (unsigned short*)alloc((size_t)4 * 9 * 256 * 256 * 2);
  unsigned short* WpHA = (unsigned short*)alloc((size_t)9 * 128 * 256 * 2);
  unsigned short* WpHB = (unsigned short*)alloc((size_t)9 * 64 * 256 * 2);
  float* stats = (float*)alloc(64 * 2 * 4);
  float* cand = (float*)alloc((size_t)2 * 5000 * 6 * 4);
  int* cnt = (int*)alloc(10 * 4);
  float* biasA = (float*)alloc(128 * 4);
  float* biasB = (float*)alloc(64 * 4);

  // weight packing + per-call init
  pack_tower_w<<<9216, 256, 0, stream>>>(cls_w, WpC);
  pack_tower_w<<<9216, 256, 0, stream>>>(box_w, WpB);
  pack_headA_w<<<1152, 256, 0, stream>>>(score_w, ctr_w, WpHA);
  pack_headB_w<<<576, 256, 0, stream>>>(bbox_w, WpHB);
  init_kernel<<<(2 * 5000 * 6 + 255) / 256, 256, 0, stream>>>(cand, cnt, out, biasA, biasB,
                                                              score_b, ctr_b, bbox_b);

  auto conv = [&](int npass, const float* src, const unsigned short* wpk, const float* bias,
                  float* dst, int H, int W, int co_store) {
    int gxt = (W + 63) / 64, gyt = (H + 1) / 2;
    dim3 grid(2 * gyt * gxt);
    if (npass == 4)
      conv3x3_wmma_kernel<4><<<grid, 256, 0, stream>>>(src, wpk, bias, dst, H, W, co_store);
    else if (npass == 2)
      conv3x3_wmma_kernel<2><<<grid, 256, 0, stream>>>(src, wpk, bias, dst, H, W, co_store);
    else
      conv3x3_wmma_kernel<1><<<grid, 256, 0, stream>>>(src, wpk, bias, dst, H, W, co_store);
  };

  for (int l = 0; l < 5; ++l) {
    const int H = Hs[l], W = Ws[l];
    // --- cls tower ---
    const float* src = feats[l];
    for (int i = 0; i < 4; ++i) {
      conv(4, src, WpC + (size_t)i * 9 * 256 * 256, cls_b + i * 256, P0, H, W, 256);
      gn_stats_kernel<<<64, 256, 0, stream>>>(P0, stats, H, W);
      gn_apply_relu_kernel<<<2048, 256, 0, stream>>>(P0, P1, stats, cls_gn_s + i * 256,
                                                     cls_gn_b + i * 256, H, W);
      src = P1;
    }
    conv(2, P1, WpHA, biasA, HA, H, W, 81);  // 80 class logits + centerness
    // --- box tower ---
    src = feats[l];
    for (int i = 0; i < 4; ++i) {
      conv(4, src, WpB + (size_t)i * 9 * 256 * 256, box_b + i * 256, P0, H, W, 256);
      gn_stats_kernel<<<64, 256, 0, stream>>>(P0, stats, H, W);
      gn_apply_relu_kernel<<<2048, 256, 0, stream>>>(P0, P1, stats, box_gn_s + i * 256,
                                                     box_gn_b + i * 256, H, W);
      src = P1;
    }
    conv(1, P1, WpHB, biasB, HB, H, W, 4);   // 4 box distances
    // --- decode this level ---
    decode_kernel<<<(2 * H * W + 255) / 256, 256, 0, stream>>>(HA, HB, scales, l, H, W, Sv[l],
                                                               cand, cnt);
  }

  nms_kernel<<<2, 256, 0, stream>>>(cand, out);
}
